// TrajlevelVectorQuantizer_64742337020153
// MI455X (gfx1250) — matrix operand order repaired and checked by reading, not compiled, not run on hardware
//
#include <hip/hip_runtime.h>
#include <hip/hip_bf16.h>
#include <stdint.h>

// Problem constants (from reference)
#define N_E    1024
#define E_DIM  512
#define B_SZ   64
#define T_SZ   512
#define N_ROWS (B_SZ * T_SZ)        // 32768 flattened z vectors
// BETA = 0.25 -> loss = 1.25 * mean((z_q - z)^2)

typedef __bf16 v16bf __attribute__((ext_vector_type(16)));
typedef __bf16 v8bf  __attribute__((ext_vector_type(8)));
typedef float  v8f   __attribute__((ext_vector_type(8)));

#if __has_builtin(__builtin_amdgcn_tensor_load_to_lds)
#define USE_TDM 1
typedef unsigned int tdm_u32x4 __attribute__((ext_vector_type(4)));
typedef int          tdm_i32x4 __attribute__((ext_vector_type(4)));
typedef int          tdm_i32x8 __attribute__((ext_vector_type(8)));
#else
#define USE_TDM 0
#endif

__device__ __forceinline__ unsigned short f32_to_bf16(float f) {
  unsigned int u = __float_as_uint(f);
  u += 0x7FFFu + ((u >> 16) & 1u);          // round to nearest even
  return (unsigned short)(u >> 16);
}

// ---------------------------------------------------------------- prep kernels

__global__ void k_zero_counts(int* __restrict__ counts) {
  if (threadIdx.x < N_E) counts[threadIdx.x] = 0;
}

// One block per code row: f32 -> bf16 + ||e||^2
__global__ void k_prep_codebook(const float* __restrict__ cb,
                                unsigned short* __restrict__ cb16,
                                float* __restrict__ normE) {
  const int e   = blockIdx.x;
  const int tid = threadIdx.x;
  const float* row = cb + (size_t)e * E_DIM;
  float s = 0.f;
  for (int c = tid; c < E_DIM; c += 256) {
    float v = row[c];
    cb16[(size_t)e * E_DIM + c] = f32_to_bf16(v);
    s += v * v;
  }
  for (int off = 16; off; off >>= 1) s += __shfl_down(s, off, 32);
  __shared__ float sred[8];
  if ((tid & 31) == 0) sred[tid >> 5] = s;
  __syncthreads();
  if (tid == 0) {
    float t = 0.f;
    for (int i = 0; i < 8; ++i) t += sred[i];
    normE[e] = t;
  }
}

// Transpose-convert z (B, C, T) f32 -> zbf16 (N=B*T, C) bf16 via 32x32 LDS tiles
__global__ void k_convert_z(const float* __restrict__ z,
                            unsigned short* __restrict__ zb) {
  __shared__ float tile[32][33];
  const int tx = threadIdx.x, ty = threadIdx.y;
  const int ct = blockIdx.x, tt = blockIdx.y, b = blockIdx.z;
  const size_t zbase = (size_t)b * E_DIM * T_SZ;
#pragma unroll
  for (int r = 0; r < 4; ++r) {
    int c = ct * 32 + ty + r * 8;
    tile[ty + r * 8][tx] = z[zbase + (size_t)c * T_SZ + tt * 32 + tx];
  }
  __syncthreads();
#pragma unroll
  for (int r = 0; r < 4; ++r) {
    int c = ct * 32 + tx;
    int t = tt * 32 + ty + r * 8;
    size_t n = (size_t)b * T_SZ + t;
    zb[n * E_DIM + c] = f32_to_bf16(tile[tx][ty + r * 8]);
  }
}

// Zero the 128MB one-hot output region (float2: out_me byte offset is 8-aligned)
__global__ void k_zero_me(float2* __restrict__ me) {
  size_t i = (size_t)blockIdx.x * blockDim.x + threadIdx.x;
  me[i] = make_float2(0.f, 0.f);
}

// ------------------------------------------------------- main WMMA argmin GEMM
// Block = 256 threads = 8 waves. Wave w owns 16 rows: TDM-staged into its own
// 16KB LDS slab, then the whole A tile is pulled into 128 VGPRs (16 x v16bf in
// WMMA operand layout) and reused across all 64 codebook column tiles. Per tile:
// 16 x v_wmma_f32_16x16x32_bf16 with a 2-stage B prefetch pipeline from the
// L2-resident bf16 codebook. Each lane keeps a per-lane running argmin of
// ||e||^2 - 2*(z.e) (its column n = ct*16 + lane&15 is ascending), with one
// cross-lane butterfly at the very end.
__global__ __launch_bounds__(256, 1)
void k_argmin_wmma(const unsigned short* __restrict__ zbf16,
                   const unsigned short* __restrict__ cb16,
                   const float* __restrict__ normE,
                   int* __restrict__ idx_i,
                   float* __restrict__ out_idx,
                   int* __restrict__ counts) {
  extern __shared__ unsigned short smem[];
  const int tid     = threadIdx.x;
  const int lane    = tid & 31;
  const int wid     = tid >> 5;
  const int rt      = blockIdx.x * 8 + wid;      // row tile id (0..2047)
  const int rowbase = rt * 16;
  unsigned short* slab = smem + wid * (16 * E_DIM);

  // ---- Stage this wave's 16x512 bf16 A tile (16KB) into LDS via the TDM ----
#if USE_TDM
  {
    const unsigned lds_off = __builtin_amdgcn_groupstaticsize()
                           + (unsigned)wid * (16u * E_DIM * 2u);
    const unsigned long long ga =
        (unsigned long long)(uintptr_t)(zbf16 + (size_t)rowbase * E_DIM);
    // D# group 0: count=1, lds_addr, 57-bit global addr, type=2 ("image")
    tdm_u32x4 g0;
    g0[0] = 1u;
    g0[1] = lds_off;
    g0[2] = (unsigned)(ga & 0xFFFFFFFFu);
    g0[3] = (unsigned)((ga >> 32) & 0x01FFFFFFu) | (2u << 30);
    // D# group 1: wg_mask=0, data_size=1 (2B), 2D tile 512x16, stride 512
    tdm_i32x8 g1;
    g1[0] = (int)(1u << 16);                               // data_size @ [17:16]
    g1[1] = (int)(((unsigned)E_DIM & 0xFFFFu) << 16);      // tensor_dim0 @ bit48
    g1[2] = (int)(16u << 16);                              // tensor_dim1 @ bit80
    g1[3] = (int)(((unsigned)E_DIM & 0xFFFFu) << 16);      // tile_dim0  @ bit112
    g1[4] = 16;                                            // tile_dim1  @ bit128
    g1[5] = E_DIM;                                         // tensor_dim0_stride @ bit160
    g1[6] = (int)(((unsigned)(16 * E_DIM) & 0xFFFFu) << 16); // tensor_dim1_stride @ bit208
    g1[7] = 0;
    tdm_i32x4 gz4 = {0, 0, 0, 0};                          // groups 2/3 unused (<=2D)
    tdm_i32x8 gz8 = {0, 0, 0, 0, 0, 0, 0, 0};              // trailing group (6-arg form)
    __builtin_amdgcn_tensor_load_to_lds(g0, g1, gz4, gz4, gz8, 0);
    __builtin_amdgcn_s_wait_tensorcnt(0);
  }
#else
  {
    const uint4* src = (const uint4*)(zbf16 + (size_t)rowbase * E_DIM);
    uint4* dst = (uint4*)slab;
    for (int i = lane; i < (16 * E_DIM) / 8; i += 32) dst[i] = src[i];
  }
#endif
  __syncthreads();

  const int m    = lane & 15;      // A row / B column within tile
  const int half = lane >> 4;      // lane half selects K sub-chunk
  // A layout (16-bit 16x32): lanes 0-15 hold K[0..7]=v0-3, K[16..23]=v4-7;
  // lanes 16-31 hold K[8..15], K[24..31]  -> chunks at kk+half*8, kk+16+half*8
  const __bf16* sA = (const __bf16*)slab + (size_t)m * E_DIM + half * 8;

  // Pull the entire A tile into registers once (16 x v16bf = 128 VGPRs),
  // pre-combined into WMMA operand layout; reused for all 64 column tiles.
  v16bf areg[16];
#pragma unroll
  for (int k = 0; k < 16; ++k) {
    v8bf a0 = *(const v8bf*)(sA + k * 32);
    v8bf a1 = *(const v8bf*)(sA + k * 32 + 16);
    areg[k] = __builtin_shufflevector(a0, a1, 0,1,2,3,4,5,6,7,8,9,10,11,12,13,14,15);
  }

  float bestd[8];
  int   besti[8];
#pragma unroll
  for (int r = 0; r < 8; ++r) { bestd[r] = 3.4e38f; besti[r] = 0; }

  for (int ct = 0; ct < N_E / 16; ++ct) {
    const int n = ct * 16 + m;
    // B layout (16-bit 32x16): lanes 0-15 hold K[0..7],K[16..23]; lanes 16-31 K[8..15],K[24..31]
    // B[k][n] = codebook[n][k] -> chunks at row n, offsets kk+half*8 and kk+16+half*8
    const __bf16* brow = (const __bf16*)cb16 + (size_t)n * E_DIM + half * 8;
    v8f acc = {0.f, 0.f, 0.f, 0.f, 0.f, 0.f, 0.f, 0.f};

    // 2-stage B prefetch pipeline
    v8bf b0 = *(const v8bf*)(brow);
    v8bf b1 = *(const v8bf*)(brow + 16);
#pragma unroll
    for (int k = 0; k < 16; ++k) {
      v8bf nb0 = b0, nb1 = b1;
      if (k < 15) {
        nb0 = *(const v8bf*)(brow + (k + 1) * 32);
        nb1 = *(const v8bf*)(brow + (k + 1) * 32 + 16);
      }
      v16bf b = __builtin_shufflevector(b0, b1, 0,1,2,3,4,5,6,7,8,9,10,11,12,13,14,15);
      acc = __builtin_amdgcn_wmma_f32_16x16x32_bf16(
          /*neg_a=*/false, areg[k], /*neg_b=*/false, b,
          /*c_mod=*/(short)0, acc, /*reuse_a=*/false, /*reuse_b=*/false);
      b0 = nb0; b1 = nb1;
    }

    // Per-lane running argmin (branchless): lane's column index n is strictly
    // increasing across ct, so strict < keeps the first (lowest) index on ties.
    const float ne = normE[n];
#pragma unroll
    for (int r = 0; r < 8; ++r) {
      float d = fmaf(-2.0f, acc[r], ne);
      bool take = d < bestd[r];
      bestd[r] = take ? d : bestd[r];
      besti[r] = take ? n : besti[r];
    }
  }

  // One cross-lane butterfly per row at the end (16 N-lanes per half)
#pragma unroll
  for (int r = 0; r < 8; ++r) {
    float d  = bestd[r];
    int   bi = besti[r];
#pragma unroll
    for (int off = 1; off < 16; off <<= 1) {
      float od = __shfl_xor(d, off, 32);
      int   oi = __shfl_xor(bi, off, 32);
      bool take = (od < d) || (od == d && oi < bi);
      d  = take ? od : d;
      bi = take ? oi : bi;
    }
    bestd[r] = d;
    besti[r] = bi;
  }

  if (m == 0) {  // lanes 0 and 16 hold rows rowbase+[0..7] and rowbase+[8..15]
#pragma unroll
    for (int r = 0; r < 8; ++r) {
      int row = rowbase + half * 8 + r;
      idx_i[row]   = besti[r];
      out_idx[row] = (float)besti[r];
      atomicAdd(&counts[besti[r]], 1);
    }
  }
}

// ------------------------------------------------ gather z_q + loss partials
__global__ void k_gather_loss(const float* __restrict__ z,
                              const float* __restrict__ cb,
                              const int* __restrict__ idx_i,
                              float* __restrict__ zq,
                              float* __restrict__ partials) {
  const int gid = blockIdx.x * blockDim.x + threadIdx.x;   // 0..4194303
  const int f4  = gid << 2;                                 // flat (b,c,t) index
  const int t   = f4 & (T_SZ - 1);
  const int c   = (f4 >> 9) & (E_DIM - 1);
  const int b   = f4 >> 18;
  const int nb  = b * T_SZ + t;
  const float4 zv = *(const float4*)(z + (size_t)f4);       // z base 16B aligned
  float q0 = cb[(size_t)idx_i[nb + 0] * E_DIM + c];
  float q1 = cb[(size_t)idx_i[nb + 1] * E_DIM + c];
  float q2 = cb[(size_t)idx_i[nb + 2] * E_DIM + c];
  float q3 = cb[(size_t)idx_i[nb + 3] * E_DIM + c];
  // z_q_ste forward value == z_q exactly; scalar stores (zq base is 4B-aligned only)
  zq[(size_t)f4 + 0] = q0;
  zq[(size_t)f4 + 1] = q1;
  zq[(size_t)f4 + 2] = q2;
  zq[(size_t)f4 + 3] = q3;
  float d0 = q0 - zv.x, d1 = q1 - zv.y, d2 = q2 - zv.z, d3 = q3 - zv.w;
  float s = d0 * d0 + d1 * d1 + d2 * d2 + d3 * d3;
  for (int off = 16; off; off >>= 1) s += __shfl_down(s, off, 32);
  __shared__ float sred[8];
  if ((threadIdx.x & 31) == 0) sred[threadIdx.x >> 5] = s;
  __syncthreads();
  if (threadIdx.x == 0) {
    float acc = 0.f;
    for (int i = 0; i < 8; ++i) acc += sred[i];
    partials[blockIdx.x] = acc;     // deterministic fixed-order reduction
  }
}

__global__ void k_scatter_onehot(const int* __restrict__ idx_i,
                                 float* __restrict__ me) {
  int n = blockIdx.x * blockDim.x + threadIdx.x;
  me[(size_t)n * N_E + idx_i[n]] = 1.0f;
}

__global__ void k_finalize(const float* __restrict__ partials, int npart,
                           const int* __restrict__ counts,
                           float* __restrict__ out_loss,
                           float* __restrict__ out_perp) {
  __shared__ float red[256];
  const int tid = threadIdx.x;
  float s = 0.f;
  for (int i = tid; i < npart; i += 256) s += partials[i];
  red[tid] = s; __syncthreads();
  for (int off = 128; off; off >>= 1) {
    if (tid < off) red[tid] += red[tid + off];
    __syncthreads();
  }
  if (tid == 0) *out_loss = 1.25f * red[0] / 16777216.0f;  // (1+beta)*mean
  __syncthreads();
  float h = 0.f;
  for (int i = tid; i < N_E; i += 256) {
    float e = (float)counts[i] / (float)N_ROWS;
    h += e * logf(e + 1e-10f);
  }
  red[tid] = h; __syncthreads();
  for (int off = 128; off; off >>= 1) {
    if (tid < off) red[tid] += red[tid + off];
    __syncthreads();
  }
  if (tid == 0) *out_perp = expf(-red[0]);
}

// --------------------------------------------------------------------- launch
extern "C" void kernel_launch(void* const* d_in, const int* in_sizes, int n_in,
                              void* d_out, int out_size, void* d_ws, size_t ws_size,
                              hipStream_t stream) {
  (void)in_sizes; (void)n_in; (void)out_size; (void)ws_size;
  const float* z  = (const float*)d_in[0];   // (64, 512, 512) f32
  const float* cb = (const float*)d_in[1];   // (1024, 512)    f32
  float* out = (float*)d_out;

  // Output tuple, concatenated flat in return order:
  float* out_loss = out;                                   // 1
  float* out_zq   = out + 1;                               // 16777216
  float* out_perp = out + 1 + 16777216;                    // 1
  float* out_me   = out + 2 + 16777216;                    // 33554432
  float* out_idx  = out + 2 + 16777216 + 33554432;         // 32768

  // Workspace layout (~34.9 MB)
  char* ws = (char*)d_ws;
  unsigned short* zbf16 = (unsigned short*)(ws + 0);           // 33,554,432 B
  unsigned short* cb16  = (unsigned short*)(ws + 33554432);    //  1,048,576 B
  float* normE    = (float*)(ws + 34603008);                   //      4,096 B
  int*   idx_i    = (int*)  (ws + 34607104);                   //    131,072 B
  int*   counts   = (int*)  (ws + 34738176);                   //      4,096 B
  float* partials = (float*)(ws + 34742272);                   //     65,536 B

  hipLaunchKernelGGL(k_zero_counts, dim3(1), dim3(1024), 0, stream, counts);
  hipLaunchKernelGGL(k_prep_codebook, dim3(N_E), dim3(256), 0, stream, cb, cb16, normE);
  hipLaunchKernelGGL(k_convert_z, dim3(16, 16, 64), dim3(32, 8), 0, stream, z, zbf16);
  hipLaunchKernelGGL(k_zero_me, dim3(65536), dim3(256), 0, stream, (float2*)out_me);
  hipLaunchKernelGGL(k_argmin_wmma, dim3(256), dim3(256),
                     8 * 16 * E_DIM * sizeof(unsigned short), stream,
                     zbf16, cb16, normE, idx_i, out_idx, counts);
  hipLaunchKernelGGL(k_gather_loss, dim3(16384), dim3(256), 0, stream,
                     z, cb, idx_i, out_zq, partials);
  hipLaunchKernelGGL(k_scatter_onehot, dim3(128), dim3(256), 0, stream, idx_i, out_me);
  hipLaunchKernelGGL(k_finalize, dim3(1), dim3(256), 0, stream,
                     partials, 16384, counts, out_loss, out_perp);
}
